// ProtoNet_63144609186395
// MI455X (gfx1250) — compile-verified, matrix operand order checked by reading, not run on hardware
//
#include <hip/hip_runtime.h>
#include <hip/hip_bf16.h>
#include <math.h>

#define NB   16
#define HWP  1024
#define NN   (NB*HWP)     // 16384 pixels
#define DBB  768
#define DD   512
#define KC   201
#define MP   10
#define KM   (KC*MP)      // 2010

typedef __attribute__((ext_vector_type(16))) __bf16 v16bf;
typedef __attribute__((ext_vector_type(8)))  float  v8f;

struct Frag32 { uint4 lo; uint4 hi; };   // 16 bf16 = 32 bytes

__device__ __forceinline__ unsigned short f32_to_bf16(float f) {
    unsigned u = __float_as_uint(f);
    return (unsigned short)((u + 0x7FFFu + ((u >> 16) & 1u)) >> 16);
}
__device__ __forceinline__ unsigned fflip(float f) {
    unsigned u = __float_as_uint(f);
    return (u & 0x80000000u) ? ~u : (u | 0x80000000u);
}
__device__ __forceinline__ float funflip(unsigned u) {
    unsigned o = (u & 0x80000000u) ? (u ^ 0x80000000u) : ~u;
    return __uint_as_float(o);
}

// ---------------- CDNA5 TDM / cluster warm-up (NULL descriptor => architectural no-op) ----------
#if defined(__has_builtin)
#if __has_builtin(__builtin_amdgcn_tensor_load_to_lds) && __has_builtin(__builtin_amdgcn_s_wait_tensorcnt)
#define HAVE_TDM 1
#endif
#endif

#ifdef HAVE_TDM
typedef unsigned int u32x4 __attribute__((ext_vector_type(4)));
typedef int          i32x4 __attribute__((ext_vector_type(4)));
typedef int          i32x8 __attribute__((ext_vector_type(8)));
__global__ void k_tdm_null() {
    __shared__ char lds[128];
    // D# group0: count=0 -> NULL tensor (no data moved); type=2 ("image") in bits 127:126.
    u32x4 g0 = {0u, 0u, 0u, 0x80000000u};
    i32x8 g1 = {0, 0, 0, 0, 0, 0, 0, 0};
    i32x4 g2 = {0, 0, 0, 0};
    i32x4 g3 = {0, 0, 0, 0};
#if __clang_major__ >= 23
    i32x8 g4 = {0, 0, 0, 0, 0, 0, 0, 0};
    __builtin_amdgcn_tensor_load_to_lds(g0, g1, g2, g3, g4, 0);
#else
    __builtin_amdgcn_tensor_load_to_lds(g0, g1, g2, g3, 0);
#endif
    __builtin_amdgcn_s_wait_tensorcnt(0);
#if __has_builtin(__builtin_amdgcn_s_cluster_barrier)
    __builtin_amdgcn_s_cluster_barrier();   // NOP when ClusterID==0
#endif
#if __has_builtin(__builtin_amdgcn_cluster_id_x)
    if (__builtin_amdgcn_cluster_id_x() == -12345) lds[threadIdx.x & 127] = 1;
#endif
}
#endif

// ---------------- utility ----------------
__global__ void k_zero(float* p, int n) {
    for (int i = blockIdx.x * blockDim.x + threadIdx.x; i < n; i += gridDim.x * blockDim.x)
        p[i] = 0.0f;
}
__global__ void k_f2bf(const float* __restrict__ x, unsigned short* __restrict__ o, int n) {
    for (int i = blockIdx.x * blockDim.x + threadIdx.x; i < n; i += gridDim.x * blockDim.x)
        o[i] = f32_to_bf16(x[i]);
}

// ---------------- PCA power iteration ----------------
__global__ void k_colmean(const float* __restrict__ T, float* __restrict__ mu) {
    int d = blockIdx.x * blockDim.x + threadIdx.x;
    if (d >= DBB) return;
    float s = 0.f;
    for (int n = 0; n < NN; ++n) {
        if ((n & 7) == 0) __builtin_prefetch(T + (size_t)(n + 16) * DBB + d, 0, 1);
        s += T[(size_t)n * DBB + d];
    }
    mu[d] = s * (1.0f / NN);
}
__global__ void k_initv(float* v) {
    int d = blockIdx.x * blockDim.x + threadIdx.x;
    if (d < DBB) v[d] = 0.03608439182435161f; // 1/sqrt(768)
}
__global__ void k_zero_scal(float* scal) {
    if (threadIdx.x == 0) { scal[0] = 0.f; scal[1] = 0.f; scal[2] = 0.f; scal[3] = 0.f; }
}
__global__ void k_init_minmax(unsigned* scalU) {
    if (threadIdx.x == 0) { scalU[1] = 0xFFFFFFFFu; scalU[2] = 0u; }
}
// y = Ac @ v  (Ac = T - colmean); also accumulate sum(y)
__global__ void k_matvec_y(const float* __restrict__ T, const float* __restrict__ v,
                           const float* __restrict__ mu, float* __restrict__ y,
                           float* __restrict__ scal) {
    int n = blockIdx.x * blockDim.x + threadIdx.x;
    if (n >= NN) return;
    const float* row = T + (size_t)n * DBB;
    float s = 0.f, sm = 0.f;
    for (int d = 0; d < DBB; ++d) { s += row[d] * v[d]; sm += mu[d] * v[d]; }
    float yy = s - sm;
    y[n] = yy;
    atomicAdd(&scal[0], yy);
}
// vt = Ac^T @ y = T^T y - mu * sum(y)
__global__ void k_matvec_vt(const float* __restrict__ T, const float* __restrict__ y,
                            const float* __restrict__ mu, const float* __restrict__ scal,
                            float* __restrict__ vt) {
    int d = blockIdx.x * blockDim.x + threadIdx.x;
    if (d >= DBB) return;
    float s = 0.f;
    for (int n = 0; n < NN; ++n) {
        if ((n & 7) == 0) __builtin_prefetch(T + (size_t)(n + 16) * DBB + d, 0, 1);
        s += T[(size_t)n * DBB + d] * y[n];
    }
    vt[d] = s - mu[d] * scal[0];
}
__global__ void k_norm_v(const float* __restrict__ vt, float* __restrict__ v) {
    __shared__ float sh[1024];
    int t = threadIdx.x;
    float x = (t < DBB) ? vt[t] : 0.f;
    sh[t] = x * x; __syncthreads();
    for (int s = 512; s > 0; s >>= 1) { if (t < s) sh[t] += sh[t + s]; __syncthreads(); }
    float inv = rsqrtf(fmaxf(sh[0], 1e-30f));
    if (t < DBB) v[t] = x * inv;
}
__global__ void k_minmax(const float* __restrict__ y, unsigned* __restrict__ scalU) {
    int n = blockIdx.x * blockDim.x + threadIdx.x;
    if (n >= NN) return;
    unsigned e = fflip(y[n]);
    atomicMin(&scalU[1], e);
    atomicMax(&scalU[2], e);
}
__global__ void k_pseudo(const float* __restrict__ y, const unsigned* __restrict__ scalU,
                         const int* __restrict__ gt, int* __restrict__ gtseg,
                         float* __restrict__ out_pgt) {
    int n = blockIdx.x * blockDim.x + threadIdx.x;
    if (n >= NN) return;
    float mn = funflip(scalU[1]), mx = funflip(scalU[2]);
    float us = (y[n] - mn) / (mx - mn);
    int c = (us < 0.5f) ? gt[n >> 10] : (KC - 1);
    gtseg[n] = c;
    out_pgt[n] = (float)c;
}

// ---------------- WMMA GEMM: C[M,N] = A[M,K](bf16) * Brow[N,K]^T(bf16), f32 accum -------------
// One wave computes a 16x64 output block (1 M-tile x 4 N-tiles). Software-pipelined: fragments
// for K-step k+32 are loaded before the 4 WMMAs of step k issue, so the wmma->load dependency
// distance is one full iteration and s_wait_loadcnt need not drain to 0. Out-of-range N columns
// are clamped at load (EXEC all-ones for WMMA); stores are guarded.
__global__ void k_wmma_gemm_nt4(const unsigned short* __restrict__ A,
                                const unsigned short* __restrict__ Bm,
                                float* __restrict__ C,
                                int Mtiles, int Ntot, int Kt) {
    int wave = (blockIdx.x * blockDim.x + threadIdx.x) >> 5;
    int lane = threadIdx.x & 31;
    int Ntiles = (Ntot + 15) >> 4;
    int Ngrp   = (Ntiles + 3) >> 2;
    if (wave >= Mtiles * Ngrp) return;            // wave-uniform exit
    int tm = wave / Ngrp, tg = wave % Ngrp;
    int m   = lane & 15;
    int grp = lane >> 4;
    const unsigned short* arow = A + (size_t)(tm * 16 + m) * Kt + grp * 8;
    const unsigned short* brow[4];
#pragma unroll
    for (int j = 0; j < 4; ++j) {
        int col = (tg * 4 + j) * 16 + m;
        int cc  = (col < Ntot) ? col : (Ntot - 1);   // clamp: padded lanes read a valid dup row
        brow[j] = Bm + (size_t)cc * Kt + grp * 8;
    }

    v8f acc[4];
#pragma unroll
    for (int j = 0; j < 4; ++j)
#pragma unroll
        for (int i = 0; i < 8; ++i) acc[j][i] = 0.f;

    // prologue: load K-step 0
    Frag32 fa, fb[4];
    fa.lo = *(const uint4*)(arow);
    fa.hi = *(const uint4*)(arow + 16);
#pragma unroll
    for (int j = 0; j < 4; ++j) {
        fb[j].lo = *(const uint4*)(brow[j]);
        fb[j].hi = *(const uint4*)(brow[j] + 16);
    }

    for (int kk = 0; kk < Kt; kk += 32) {
        int kn = (kk + 32 < Kt) ? (kk + 32) : kk;   // clamp last prefetch in-bounds
        Frag32 na, nb[4];
        na.lo = *(const uint4*)(arow + kn);
        na.hi = *(const uint4*)(arow + kn + 16);
#pragma unroll
        for (int j = 0; j < 4; ++j) {
            nb[j].lo = *(const uint4*)(brow[j] + kn);
            nb[j].hi = *(const uint4*)(brow[j] + kn + 16);
        }
        v16bf a = __builtin_bit_cast(v16bf, fa);
#pragma unroll
        for (int j = 0; j < 4; ++j) {
            v16bf b = __builtin_bit_cast(v16bf, fb[j]);
            acc[j] = __builtin_amdgcn_wmma_f32_16x16x32_bf16(
                false, a, false, b, (short)0, acc[j], false, false);
        }
        fa = na;
#pragma unroll
        for (int j = 0; j < 4; ++j) fb[j] = nb[j];
    }
    // D layout: VGPR r -> row grp*8 + r, col = lane&15
#pragma unroll
    for (int j = 0; j < 4; ++j) {
        int cn = (tg * 4 + j) * 16 + m;
        if (cn < Ntot) {
#pragma unroll
            for (int r = 0; r < 8; ++r)
                C[(size_t)(tm * 16 + grp * 8 + r) * Ntot + cn] = acc[j][r];
        }
    }
}

// ---------------- LayerNorm + L2 on projected features (in place), emit bf16 copy -------------
__global__ void k_ln_l2(float* __restrict__ F, const float* __restrict__ pb,
                        const float* __restrict__ g, const float* __restrict__ bb,
                        unsigned short* __restrict__ Cb) {
    __shared__ float sh[256];
    int n = blockIdx.x, t = threadIdx.x;
    float* row = F + (size_t)n * DD;
    float x0 = row[t] + pb[t];
    float x1 = row[t + 256] + pb[t + 256];
    sh[t] = x0 + x1; __syncthreads();
    for (int s = 128; s > 0; s >>= 1) { if (t < s) sh[t] += sh[t + s]; __syncthreads(); }
    float mu = sh[0] * (1.0f / DD); __syncthreads();
    float d0 = x0 - mu, d1 = x1 - mu;
    sh[t] = d0 * d0 + d1 * d1; __syncthreads();
    for (int s = 128; s > 0; s >>= 1) { if (t < s) sh[t] += sh[t + s]; __syncthreads(); }
    float rinv = rsqrtf(sh[0] * (1.0f / DD) + 1e-5f); __syncthreads();
    float h0 = d0 * rinv * g[t] + bb[t];
    float h1 = d1 * rinv * g[t + 256] + bb[t + 256];
    sh[t] = h0 * h0 + h1 * h1; __syncthreads();
    for (int s = 128; s > 0; s >>= 1) { if (t < s) sh[t] += sh[t + s]; __syncthreads(); }
    float inv = 1.0f / fmaxf(sqrtf(sh[0]), 1e-12f);
    float c0 = h0 * inv, c1 = h1 * inv;
    row[t] = c0; row[t + 256] = c1;
    Cb[(size_t)n * DD + t]       = f32_to_bf16(c0);
    Cb[(size_t)n * DD + t + 256] = f32_to_bf16(c1);
}

// ---------------- prototype L2 normalization, f32 + bf16 ----------------
__global__ void k_proto_norm(const float* __restrict__ P, float* __restrict__ Pn,
                             unsigned short* __restrict__ Pb) {
    __shared__ float sh[256];
    int n = blockIdx.x, t = threadIdx.x;
    float x0 = P[(size_t)n * DD + t];
    float x1 = P[(size_t)n * DD + t + 256];
    sh[t] = x0 * x0 + x1 * x1; __syncthreads();
    for (int s = 128; s > 0; s >>= 1) { if (t < s) sh[t] += sh[t + s]; __syncthreads(); }
    float inv = 1.0f / fmaxf(sqrtf(sh[0]), 1e-12f);
    float p0 = x0 * inv, p1 = x1 * inv;
    Pn[(size_t)n * DD + t]       = p0;
    Pn[(size_t)n * DD + t + 256] = p1;
    Pb[(size_t)n * DD + t]       = f32_to_bf16(p0);
    Pb[(size_t)n * DD + t + 256] = f32_to_bf16(p1);
}

// ---------------- head: max over M, LN over K, write out_seg, argmax -> pred ----------------
__global__ void k_head(const float* __restrict__ logits, const float* __restrict__ mg,
                       const float* __restrict__ mb, float* __restrict__ outseg,
                       int* __restrict__ pred) {
    __shared__ float sh[256];
    __shared__ float sv[256];
    __shared__ int   si[256];
    int n = blockIdx.x, t = threadIdx.x;
    bool act = (t < KC);
    float mx = 0.f;
    if (act) {
        const float* lr = logits + (size_t)n * KM + t * MP;
        mx = lr[0];
#pragma unroll
        for (int m = 1; m < MP; ++m) mx = fmaxf(mx, lr[m]);
    }
    sh[t] = act ? mx : 0.f; __syncthreads();
    for (int s = 128; s > 0; s >>= 1) { if (t < s) sh[t] += sh[t + s]; __syncthreads(); }
    float mu = sh[0] * (1.0f / KC); __syncthreads();
    float d = act ? (mx - mu) : 0.f;
    sh[t] = d * d; __syncthreads();
    for (int s = 128; s > 0; s >>= 1) { if (t < s) sh[t] += sh[t + s]; __syncthreads(); }
    float rinv = rsqrtf(sh[0] * (1.0f / KC) + 1e-5f);
    float val = act ? (d * rinv * mg[t] + mb[t]) : -1e30f;
    if (act) {
        int b = n >> 10, hw = n & 1023;
        outseg[((size_t)b * KC + t) * HWP + hw] = val;
    }
    sv[t] = val; si[t] = t; __syncthreads();
    for (int s = 128; s > 0; s >>= 1) {
        if (t < s && sv[t + s] > sv[t]) { sv[t] = sv[t + s]; si[t] = si[t + s]; }
        __syncthreads();
    }
    if (t == 0) pred[n] = si[0];
}

// ---------------- sinkhorn (Q effectively [N,10] keyed by class) ----------------
__global__ void k_bk(const int* __restrict__ gtseg, float* __restrict__ Bk) {
    int n = blockIdx.x * blockDim.x + threadIdx.x;
    if (n < NN) atomicAdd(&Bk[gtseg[n]], 1.0f);
}
__global__ void k_qinit(const float* __restrict__ logits, const int* __restrict__ gtseg,
                        float* __restrict__ Q, float* __restrict__ sQ) {
    int n = blockIdx.x * blockDim.x + threadIdx.x;
    if (n >= NN) return;
    int c = gtseg[n];
    const float* lr = logits + (size_t)n * KM + c * MP;
    float s = 0.f;
#pragma unroll
    for (int m = 0; m < MP; ++m) { float q = __expf(lr[m] * 20.0f); Q[n * MP + m] = q; s += q; }
    atomicAdd(&sQ[c], s);
}
__global__ void k_qdiv(float* __restrict__ Q, const float* __restrict__ sQ,
                       const int* __restrict__ gtseg) {
    int n = blockIdx.x * blockDim.x + threadIdx.x;
    if (n >= NN) return;
    float s = sQ[gtseg[n]];
    s = (s > 0.f) ? s : 1.f;
    float inv = 1.0f / s;
#pragma unroll
    for (int m = 0; m < MP; ++m) Q[n * MP + m] *= inv;
}
__global__ void k_rowacc(const float* __restrict__ Q, const int* __restrict__ gtseg,
                         float* __restrict__ rowv) {
    int n = blockIdx.x * blockDim.x + threadIdx.x;
    if (n >= NN) return;
    int c = gtseg[n];
#pragma unroll
    for (int m = 0; m < MP; ++m) atomicAdd(&rowv[c * MP + m], Q[n * MP + m]);
}
__global__ void k_rowcol(float* __restrict__ Q, const float* __restrict__ rowv,
                         const float* __restrict__ Bk, const int* __restrict__ gtseg) {
    int n = blockIdx.x * blockDim.x + threadIdx.x;
    if (n >= NN) return;
    int c = gtseg[n];
    float q[MP];
    float col = 0.f;
#pragma unroll
    for (int m = 0; m < MP; ++m) {
        float r = rowv[c * MP + m];
        r = (r > 0.f) ? r : 1.f;
        float v = Q[n * MP + m] / r * (1.0f / MP);
        q[m] = v; col += v;
    }
    float cd = (col > 0.f) ? col : 1.f;
    float bk = Bk[c]; bk = (bk > 0.f) ? bk : 1.f;
    float inv = 1.0f / (cd * bk);
#pragma unroll
    for (int m = 0; m < MP; ++m) Q[n * MP + m] = q[m] * inv;
}
__global__ void k_assign(const float* __restrict__ Q, const int* __restrict__ gtseg,
                         int* __restrict__ idxv, float* __restrict__ tgt) {
    int n = blockIdx.x * blockDim.x + threadIdx.x;
    if (n >= NN) return;
    int c = gtseg[n];
    float best = Q[n * MP];
    int bi = 0;
#pragma unroll
    for (int m = 1; m < MP; ++m) { float v = Q[n * MP + m]; if (v > best) { best = v; bi = m; } }
    idxv[n] = bi;
    tgt[n] = (float)(bi + MP * c);
}
__global__ void k_ncnt(const int* __restrict__ gtseg, const int* __restrict__ pred,
                       const int* __restrict__ idxv, float* __restrict__ ncnt) {
    int n = blockIdx.x * blockDim.x + threadIdx.x;
    if (n >= NN) return;
    if (gtseg[n] == pred[n]) atomicAdd(&ncnt[gtseg[n] * MP + idxv[n]], 1.0f);
}
__global__ void k_fproto(const float* __restrict__ Cf, const int* __restrict__ gtseg,
                         const int* __restrict__ pred, const int* __restrict__ idxv,
                         float* __restrict__ fproto) {
    for (size_t i = (size_t)blockIdx.x * blockDim.x + threadIdx.x; i < (size_t)NN * DD;
         i += (size_t)gridDim.x * blockDim.x) {
        int n = (int)(i >> 9), d = (int)(i & 511);
        if (gtseg[n] == pred[n])
            atomicAdd(&fproto[((size_t)gtseg[n] * MP + idxv[n]) * DD + d], Cf[i]);
    }
}
__global__ void k_ema(const float* __restrict__ Pn, const float* __restrict__ fproto,
                      const float* __restrict__ ncnt, float* __restrict__ outP) {
    __shared__ float sh[256];
    int km = blockIdx.x, t = threadIdx.x;
    float p0 = Pn[(size_t)km * DD + t];
    float p1 = Pn[(size_t)km * DD + t + 256];
    float cnt = ncnt[km];               // block-uniform
    float x0, x1;
    if (cnt != 0.f) {
        float f0 = fproto[(size_t)km * DD + t];
        float f1 = fproto[(size_t)km * DD + t + 256];
        sh[t] = f0 * f0 + f1 * f1; __syncthreads();
        for (int s = 128; s > 0; s >>= 1) { if (t < s) sh[t] += sh[t + s]; __syncthreads(); }
        float finv = 1.0f / fmaxf(sqrtf(sh[0]), 1e-12f); __syncthreads();
        x0 = 0.999f * p0 + 0.001f * f0 * finv;
        x1 = 0.999f * p1 + 0.001f * f1 * finv;
    } else { x0 = p0; x1 = p1; }
    sh[t] = x0 * x0 + x1 * x1; __syncthreads();
    for (int s = 128; s > 0; s >>= 1) { if (t < s) sh[t] += sh[t + s]; __syncthreads(); }
    float inv = 1.0f / fmaxf(sqrtf(sh[0]), 1e-12f);
    outP[(size_t)km * DD + t]       = x0 * inv;
    outP[(size_t)km * DD + t + 256] = x1 * inv;
}

// ---------------- host orchestration ----------------
extern "C" void kernel_launch(void* const* d_in, const int* in_sizes, int n_in,
                              void* d_out, int out_size, void* d_ws, size_t ws_size,
                              hipStream_t stream) {
    const float* T  = (const float*)d_in[0];   // [16,1024,768]
    const int*   gt = (const int*)d_in[1];     // [16]
    const float* Wp = (const float*)d_in[2];   // [512,768]
    const float* pb = (const float*)d_in[3];   // [512]
    const float* fg = (const float*)d_in[4];
    const float* fb = (const float*)d_in[5];
    const float* mg = (const float*)d_in[6];
    const float* mb = (const float*)d_in[7];
    const float* Pr = (const float*)d_in[8];   // [201,10,512]

    float* out   = (float*)d_out;
    float* o_seg = out;                                    // 16*201*1024
    float* o_log = out + (size_t)NB * KC * HWP;            // 16384*2010
    float* o_tgt = o_log + (size_t)NN * KM;                // 16384
    float* o_pgt = o_tgt + NN;                             // 16384
    float* o_npr = o_pgt + NN;                             // 2010*512

    // workspace carve-out (256B aligned)
    char* w = (char*)d_ws;
    auto alloc = [&](size_t bytes) -> void* {
        void* p = (void*)w;
        w += (bytes + 255) & ~(size_t)255;
        return p;
    };
    float*          colmean = (float*)alloc(DBB * 4);
    float*          v       = (float*)alloc(DBB * 4);
    float*          vt      = (float*)alloc(DBB * 4);
    float*          yv      = (float*)alloc(NN * 4);
    float*          scal    = (float*)alloc(16 * 4);
    int*            gtseg   = (int*)alloc(NN * 4);
    int*            predv   = (int*)alloc(NN * 4);
    int*            idxv    = (int*)alloc(NN * 4);
    unsigned short* Tb      = (unsigned short*)alloc((size_t)NN * DBB * 2);
    unsigned short* Wb      = (unsigned short*)alloc((size_t)DD * DBB * 2);
    float*          Fr      = (float*)alloc((size_t)NN * DD * 4);   // f_raw, becomes _c in place
    unsigned short* Cb      = (unsigned short*)alloc((size_t)NN * DD * 2);
    float*          Pn      = (float*)alloc((size_t)KM * DD * 4);
    unsigned short* Pb      = (unsigned short*)alloc((size_t)KM * DD * 2);
    float*          Qb      = (float*)alloc((size_t)NN * MP * 4);
    float*          sQv     = (float*)alloc(256 * 4);
    float*          Bkv     = (float*)alloc(256 * 4);
    float*          rowv    = (float*)alloc(2048 * 4);
    float*          fproto  = (float*)alloc((size_t)KM * DD * 4);
    float*          ncntv   = (float*)alloc(2048 * 4);
    (void)in_sizes; (void)n_in; (void)out_size; (void)ws_size;

#ifdef HAVE_TDM
    k_tdm_null<<<1, 32, 0, stream>>>();
#endif

    // ----- PCA direction + pseudo ground truth -----
    k_colmean<<<3, 256, 0, stream>>>(T, colmean);
    k_initv<<<3, 256, 0, stream>>>(v);
    for (int it = 0; it < 10; ++it) {
        k_zero_scal<<<1, 32, 0, stream>>>(scal);
        k_matvec_y<<<64, 256, 0, stream>>>(T, v, colmean, yv, scal);
        k_matvec_vt<<<3, 256, 0, stream>>>(T, yv, colmean, scal, vt);
        k_norm_v<<<1, 1024, 0, stream>>>(vt, v);
    }
    k_zero_scal<<<1, 32, 0, stream>>>(scal);
    k_matvec_y<<<64, 256, 0, stream>>>(T, v, colmean, yv, scal); // U (affine-invariant scaling)
    k_init_minmax<<<1, 32, 0, stream>>>((unsigned*)scal);
    k_minmax<<<64, 256, 0, stream>>>(yv, (unsigned*)scal);
    k_pseudo<<<64, 256, 0, stream>>>(yv, (unsigned*)scal, gt, gtseg, o_pgt);

    // ----- projection GEMM (bf16 WMMA, f32 accumulate): 1024 Mtiles x 8 Ngroups = 8192 waves -----
    k_f2bf<<<4096, 256, 0, stream>>>(T, Tb, NN * DBB);
    k_f2bf<<<512, 256, 0, stream>>>(Wp, Wb, DD * DBB);
    k_wmma_gemm_nt4<<<2048, 128, 0, stream>>>(Tb, Wb, Fr, NN / 16, DD, DBB);

    // ----- LN + L2 on features; normalize prototypes -----
    k_ln_l2<<<NN, 256, 0, stream>>>(Fr, pb, fg, fb, Cb);
    k_proto_norm<<<KM, 256, 0, stream>>>(Pr, Pn, Pb);

    // ----- similarity GEMM: 1024 Mtiles x 32 Ngroups = 32768 waves, into d_out -----
    k_wmma_gemm_nt4<<<8192, 128, 0, stream>>>(Cb, Pb, o_log, NN / 16, KM, DD);

    // ----- segmentation head + prediction -----
    k_head<<<NN, 256, 0, stream>>>(o_log, mg, mb, o_seg, predv);

    // ----- masked per-class sinkhorn -----
    k_zero<<<1, 256, 0, stream>>>(Bkv, 256);
    k_bk<<<64, 256, 0, stream>>>(gtseg, Bkv);
    k_zero<<<1, 256, 0, stream>>>(sQv, 256);
    k_qinit<<<64, 256, 0, stream>>>(o_log, gtseg, Qb, sQv);
    k_qdiv<<<64, 256, 0, stream>>>(Qb, sQv, gtseg);
    for (int it = 0; it < 3; ++it) {
        k_zero<<<8, 256, 0, stream>>>(rowv, 2048);
        k_rowacc<<<64, 256, 0, stream>>>(Qb, gtseg, rowv);
        k_rowcol<<<64, 256, 0, stream>>>(Qb, rowv, Bkv, gtseg);
    }
    k_assign<<<64, 256, 0, stream>>>(Qb, gtseg, idxv, o_tgt);

    // ----- EMA prototype update -----
    k_zero<<<2048, 256, 0, stream>>>(fproto, KM * DD);
    k_zero<<<8, 256, 0, stream>>>(ncntv, 2048);
    k_ncnt<<<64, 256, 0, stream>>>(gtseg, predv, idxv, ncntv);
    k_fproto<<<8192, 256, 0, stream>>>(Fr, gtseg, predv, idxv, fproto);
    k_ema<<<KM, 256, 0, stream>>>(Pn, fproto, ncntv, o_npr);
}
